// _IPEXAttentionRef_65111704207718
// MI455X (gfx1250) — matrix operand order repaired and checked by reading, not compiled, not run on hardware
//
#include <hip/hip_runtime.h>

// Problem constants (match reference)
#define BB   2
#define SS   2048
#define HHID 4096
#define NHD  32
#define HDD  128
#define N3   12288               // 3*H
#define KDIM 4096                // GEMM K
#define MDIM 4096                // B*S
#define ATT_SCALE 0.08838834764831845f  // 1/sqrt(128)

#if defined(__has_builtin)
#if __has_builtin(__builtin_amdgcn_global_load_async_to_lds_b128)
#define HAVE_ASYNC_LDS 1
#endif
#endif

typedef __attribute__((ext_vector_type(16))) __bf16 v16bf;
typedef __attribute__((ext_vector_type(8)))  float  v8f;
typedef __attribute__((ext_vector_type(4)))  int    v4i;

// ---------------- bf16 helpers ----------------
static __device__ __forceinline__ unsigned short bfbits(float f) {
  union { float f; unsigned u; } x; x.f = f;
  unsigned r = x.u + 0x7fffu + ((x.u >> 16) & 1u);   // round-to-nearest-even
  return (unsigned short)(r >> 16);
}
static __device__ __forceinline__ __bf16 f2bf(float f) {
  unsigned short h = bfbits(f);
  return __builtin_bit_cast(__bf16, h);
}
static __device__ __forceinline__ float bf2f(__bf16 x) {
  unsigned short h = __builtin_bit_cast(unsigned short, x);
  union { unsigned u; float f; } t; t.u = ((unsigned)h) << 16;
  return t.f;
}

union FragU { uint4 u[2]; v16bf v; };

// A-operand fragment: 16(M) x 32(K) tile, row-major with leading dim ld.
// ISA layout: lane(0..15)=M; halves h -> K = 16*(h>>3) + 8*hi + (h&7)
static __device__ __forceinline__ v16bf frag_a(const __bf16* base, int ld) {
  int lane = threadIdx.x & 31;
  int m  = lane & 15;
  int kb = (lane >> 4) << 3;          // 0 or 8
  FragU t;
  t.u[0] = *(const uint4*)(base + (size_t)m * ld + kb);
  t.u[1] = *(const uint4*)(base + (size_t)m * ld + 16 + kb);
  return t.v;
}
// B-operand fragment: 32(K) x 16(N) tile given as row-major N x K (ld = K stride).
// ISA layout: lane(0..15)=N; halves h -> K = 16*(h>>3) + 8*hi + (h&7)
static __device__ __forceinline__ v16bf frag_b(const __bf16* base, int ld) {
  int lane = threadIdx.x & 31;
  int n  = lane & 15;
  int kb = (lane >> 4) << 3;          // 0 or 8
  FragU t;
  t.u[0] = *(const uint4*)(base + (size_t)n * ld + kb);
  t.u[1] = *(const uint4*)(base + (size_t)n * ld + 16 + kb);
  return t.v;
}

#ifdef HAVE_ASYNC_LDS
typedef __attribute__((address_space(1))) v4i* as1_v4i;
typedef __attribute__((address_space(3))) v4i* as3_v4i;
// copy 16 bf16 (32B) global -> LDS via the CDNA5 async path (ASYNCcnt-tracked)
static __device__ __forceinline__ void async_cp16(const __bf16* g, __bf16* l) {
  as1_v4i gp = (as1_v4i)((void*)g);
  as3_v4i lp = (as3_v4i)((void*)l);
  __builtin_amdgcn_global_load_async_to_lds_b128(gp,     lp,     0, 0);
  __builtin_amdgcn_global_load_async_to_lds_b128(gp + 1, lp + 1, 0, 0);
}
static __device__ __forceinline__ void wait_async0() {
#if __has_builtin(__builtin_amdgcn_s_wait_asynccnt)
  __builtin_amdgcn_s_wait_asynccnt(0);
#else
  asm volatile("s_wait_asynccnt 0x0" ::: "memory");
#endif
}
#endif

// ---------------- kernel 1: fp32 -> bf16 convert (8 elems/thread) ----------------
__global__ void __launch_bounds__(256) cvt_bf16_kernel(const float* __restrict__ src,
                                                       __bf16* __restrict__ dst, int n8) {
  int i = blockIdx.x * blockDim.x + threadIdx.x;
  if (i >= n8) return;
  const float4* s = (const float4*)(src + (size_t)i * 8);
  float4 a = s[0], b = s[1];
  union { unsigned short us[8]; uint4 u; } o;
  o.us[0] = bfbits(a.x); o.us[1] = bfbits(a.y); o.us[2] = bfbits(a.z); o.us[3] = bfbits(a.w);
  o.us[4] = bfbits(b.x); o.us[5] = bfbits(b.y); o.us[6] = bfbits(b.z); o.us[7] = bfbits(b.w);
  *(uint4*)(dst + (size_t)i * 8) = o.u;
}

// ---------------- kernel 2: QKV GEMM  P[M,12288] = X[M,K] * W[12288,K]^T ----------------
// 8 waves tiled 4(M) x 2(N); each wave owns 32x64 output (2x4 WMMA tiles).
// Double-buffered LDS, async global->LDS copies when available.
__global__ void __launch_bounds__(256) qkv_gemm_kernel(const __bf16* __restrict__ X,
                                                       const __bf16* __restrict__ W,
                                                       __bf16* __restrict__ P) {
  __shared__ __bf16 lX[2][128 * 32];
  __shared__ __bf16 lW[2][128 * 32];
  const int tid  = threadIdx.x;
  const int wv   = tid >> 5;
  const int lane = tid & 31;
  const int mw = wv >> 1;              // 0..3 : M wave coord
  const int nw = wv & 1;               // 0..1 : N wave coord
  const int m0 = blockIdx.y * 128;
  const int n0 = blockIdx.x * 128;

  v8f acc[2][4] = {};

  const int row = tid >> 1;
  const int cb  = (tid & 1) * 16;
  const __bf16* gX = X + (size_t)(m0 + row) * KDIM + cb;
  const __bf16* gW = W + (size_t)(n0 + row) * KDIM + cb;
  __bf16* dXa = &lX[0][row * 32 + cb];
  __bf16* dWa = &lW[0][row * 32 + cb];
  __bf16* dXb = &lX[1][row * 32 + cb];
  __bf16* dWb = &lW[1][row * 32 + cb];
  const int NT = KDIM / 32;

#ifdef HAVE_ASYNC_LDS
  // -------- async double-buffered pipeline --------
  async_cp16(gX, dXa);
  async_cp16(gW, dWa);
  for (int t = 0; t < NT; ++t) {
    wait_async0();
    __syncthreads();
    if (t + 1 < NT) {
      int k1 = (t + 1) * 32;
      async_cp16(gX + k1, ((t + 1) & 1) ? dXb : dXa);
      async_cp16(gW + k1, ((t + 1) & 1) ? dWb : dWa);
    }
    const __bf16* bx = &lX[t & 1][0];
    const __bf16* bw = &lW[t & 1][0];
    v16bf af[2], bf[4];
#pragma unroll
    for (int mi = 0; mi < 2; ++mi) af[mi] = frag_a(bx + (mw * 32 + mi * 16) * 32, 32);
#pragma unroll
    for (int ni = 0; ni < 4; ++ni) bf[ni] = frag_b(bw + (nw * 64 + ni * 16) * 32, 32);
#pragma unroll
    for (int mi = 0; mi < 2; ++mi)
#pragma unroll
      for (int ni = 0; ni < 4; ++ni)
        acc[mi][ni] = __builtin_amdgcn_wmma_f32_16x16x32_bf16(false, af[mi], false, bf[ni],
                                                              (short)0, acc[mi][ni], false, false);
  }
#else
  // -------- synchronous fallback (single buffer) --------
  for (int t = 0; t < NT; ++t) {
    int k0 = t * 32;
    uint4 x0 = *(const uint4*)(gX + k0), x1 = *(const uint4*)(gX + k0 + 8);
    uint4 w0 = *(const uint4*)(gW + k0), w1 = *(const uint4*)(gW + k0 + 8);
    *(uint4*)dXa = x0; *(uint4*)(dXa + 8) = x1;
    *(uint4*)dWa = w0; *(uint4*)(dWa + 8) = w1;
    __syncthreads();
    v16bf af[2], bf[4];
#pragma unroll
    for (int mi = 0; mi < 2; ++mi) af[mi] = frag_a(&lX[0][(mw * 32 + mi * 16) * 32], 32);
#pragma unroll
    for (int ni = 0; ni < 4; ++ni) bf[ni] = frag_b(&lW[0][(nw * 64 + ni * 16) * 32], 32);
#pragma unroll
    for (int mi = 0; mi < 2; ++mi)
#pragma unroll
      for (int ni = 0; ni < 4; ++ni)
        acc[mi][ni] = __builtin_amdgcn_wmma_f32_16x16x32_bf16(false, af[mi], false, bf[ni],
                                                              (short)0, acc[mi][ni], false, false);
    __syncthreads();
  }
#endif

  // epilogue: D layout M = r + 8*hi, N = lane&15
  const int nn = lane & 15, hi = lane >> 4;
#pragma unroll
  for (int mi = 0; mi < 2; ++mi)
#pragma unroll
    for (int ni = 0; ni < 4; ++ni)
#pragma unroll
      for (int r = 0; r < 8; ++r) {
        size_t mrow = (size_t)(m0 + mw * 32 + mi * 16 + r + 8 * hi);
        P[mrow * N3 + n0 + nw * 64 + ni * 16 + nn] = f2bf(acc[mi][ni][r]);
      }
}

// ---------------- kernel 3: RoPE + reshape (Q,K -> [B,NH,S,HD]; V -> [B,NH,HD,S]) -----
__global__ void __launch_bounds__(256) rope_kernel(const __bf16* __restrict__ Pr,
                                                   __bf16* __restrict__ Qr,
                                                   __bf16* __restrict__ Kr,
                                                   __bf16* __restrict__ Vt) {
  int idx = blockIdx.x * blockDim.x + threadIdx.x;   // B*S*NH*64 threads
  int d = idx & 63;
  int h = (idx >> 6) & 31;
  int s = (idx >> 11) & 2047;
  int b = idx >> 22;

  size_t src = ((size_t)(b * SS + s)) * N3 + h * HDD;
  float q1 = bf2f(Pr[src + d]),        q2 = bf2f(Pr[src + d + 64]);
  float k1 = bf2f(Pr[src + 4096 + d]), k2 = bf2f(Pr[src + 4096 + d + 64]);
  float v1 = bf2f(Pr[src + 8192 + d]), v2 = bf2f(Pr[src + 8192 + d + 64]);

  // inv_freq = 10000^(-d/64); ln(10000)/64 = 0.14391156831212787
  float ang = (float)s * __expf(-(float)d * 0.14391156831212787f);
  float c = __cosf(ang), sn = __sinf(ang);

  size_t bh = (size_t)(b * NHD + h);
  size_t dq = (bh * SS + s) * HDD + d;
  Qr[dq]      = f2bf(q1 * c - q2 * sn);
  Qr[dq + 64] = f2bf(q2 * c + q1 * sn);
  Kr[dq]      = f2bf(k1 * c - k2 * sn);
  Kr[dq + 64] = f2bf(k2 * c + k1 * sn);

  size_t dv = (bh * HDD + d) * SS + s;               // transposed V
  Vt[dv]           = f2bf(v1);
  Vt[dv + 64 * SS] = f2bf(v2);
}

// ---------------- kernel 4: causal flash attention --------------------------------
// block = 8 waves; each wave owns 16 query rows; online softmax; 32-wide KV tiles.
__global__ void __launch_bounds__(256) attn_kernel(const __bf16* __restrict__ Q,
                                                   const __bf16* __restrict__ Kc,
                                                   const __bf16* __restrict__ Vt,
                                                   float* __restrict__ out) {
  __shared__ float  sS[8][16][34];   // scores (padded vs bank conflicts)
  __shared__ __bf16 sP[8][16][32];   // exp'd probs, row-major 16x32
  __shared__ float  sR[8][16];       // per-row rescale
  __shared__ float  sL[8][16];       // per-row 1/l

  const int tid  = threadIdx.x;
  const int wv   = tid >> 5;
  const int lane = tid & 31;
  const int nn = lane & 15, hi = lane >> 4;

  const int h = blockIdx.y, b = blockIdx.z;
  const int q0 = blockIdx.x * 128 + wv * 16;
  const size_t bh = (size_t)(b * NHD + h);
  const __bf16* Qb = Q  + (bh * SS + q0) * HDD;
  const __bf16* Kb = Kc + bh * SS * HDD;
  const __bf16* Vb = Vt + bh * HDD * SS;

  // preload Q fragments (16 x 128 = 4 K-chunks)
  v16bf qf[4];
#pragma unroll
  for (int kk = 0; kk < 4; ++kk) qf[kk] = frag_a(Qb + kk * 32, HDD);

  // ctx^T accumulators: 8 chunks of 16(d) x 16(q); lane = q, reg = d
  v8f ctx[8] = {};
  float m_run = -1e30f, l_run = 0.f;

  const int ntiles = (q0 + 16 + 31) >> 5;
  for (int t = 0; t < ntiles; ++t) {
    const int j0 = t * 32;

    // prefetch next KV tile (lowers to global_prefetch_b8)
    if (t + 1 < ntiles) {
      __builtin_prefetch(Kb + (size_t)(j0 + 32 + lane) * HDD, 0, 1);
      __builtin_prefetch(Vb + (size_t)(lane * 4) * SS + j0 + 32, 0, 1);
    }

    // ---- scores S = Q K^T for two 16-wide kv subtiles; preload all 8 K frags ----
    v16bf kf0[4], kf1[4];
#pragma unroll
    for (int kk = 0; kk < 4; ++kk) {
      kf0[kk] = frag_b(Kb + (size_t)(j0)      * HDD + kk * 32, HDD);
      kf1[kk] = frag_b(Kb + (size_t)(j0 + 16) * HDD + kk * 32, HDD);
    }
    v8f s0 = {}, s1 = {};
#pragma unroll
    for (int kk = 0; kk < 4; ++kk) {   // two independent chains interleave
      s0 = __builtin_amdgcn_wmma_f32_16x16x32_bf16(false, qf[kk], false, kf0[kk],
                                                   (short)0, s0, false, false);
      s1 = __builtin_amdgcn_wmma_f32_16x16x32_bf16(false, qf[kk], false, kf1[kk],
                                                   (short)0, s1, false, false);
    }
#pragma unroll
    for (int r = 0; r < 8; ++r) {
      int row = r + 8 * hi;
      float v0 = s0[r] * ATT_SCALE, v1 = s1[r] * ATT_SCALE;
      sS[wv][row][nn]      = (j0 + nn      <= q0 + row) ? v0 : -1e30f;
      sS[wv][row][16 + nn] = (j0 + 16 + nn <= q0 + row) ? v1 : -1e30f;
    }
    asm volatile("s_wait_dscnt 0x0" ::: "memory");

    // ---- online softmax: lane la (<16) owns query row la ----
    if (lane < 16) {
      float tmax = -1e30f;
#pragma unroll
      for (int j = 0; j < 32; ++j) tmax = fmaxf(tmax, sS[wv][lane][j]);
      float m_new = fmaxf(m_run, tmax);
      float resc  = __expf(m_run - m_new);
      float lsum  = 0.f;
#pragma unroll
      for (int j = 0; j < 32; ++j) {
        float p = __expf(sS[wv][lane][j] - m_new);
        lsum += p;
        sP[wv][lane][j] = f2bf(p);
      }
      l_run = l_run * resc + lsum;
      m_run = m_new;
      sR[wv][lane] = resc;
    }
    asm volatile("s_wait_dscnt 0x0" ::: "memory");

    // ---- rescale accumulated ctx (per-lane factor: lane = query) ----
    float f = sR[wv][nn];
#pragma unroll
    for (int c = 0; c < 8; ++c)
#pragma unroll
      for (int e = 0; e < 8; ++e) ctx[c][e] *= f;

    // ---- ctx^T += V^T P^T : preload all 8 V frags, then WMMA burst ----
    v16bf pf = frag_b(&sP[wv][0][0], 32);
    v16bf vf[8];
#pragma unroll
    for (int c = 0; c < 8; ++c) vf[c] = frag_a(Vb + (size_t)(c * 16) * SS + j0, SS);
#pragma unroll
    for (int c = 0; c < 8; ++c)
      ctx[c] = __builtin_amdgcn_wmma_f32_16x16x32_bf16(false, vf[c], false, pf,
                                                       (short)0, ctx[c], false, false);
  }

  // ---- finalize: divide by l, write out [B,S,H] ----
  if (lane < 16) sL[wv][lane] = 1.f / l_run;
  asm volatile("s_wait_dscnt 0x0" ::: "memory");
  float linv = sL[wv][nn];

  float* ob = out + ((size_t)b * SS + q0 + nn) * HHID + h * HDD + 8 * hi;
#pragma unroll
  for (int c = 0; c < 8; ++c) {
    union { v8f v; float4 f4[2]; } u;
#pragma unroll
    for (int e = 0; e < 8; ++e) u.v[e] = ctx[c][e] * linv;
    *(float4*)(ob + c * 16)     = u.f4[0];
    *(float4*)(ob + c * 16 + 4) = u.f4[1];
  }
}

// ---------------- launch ----------------
extern "C" void kernel_launch(void* const* d_in, const int* in_sizes, int n_in,
                              void* d_out, int out_size, void* d_ws, size_t ws_size,
                              hipStream_t stream) {
  (void)in_sizes; (void)n_in; (void)out_size; (void)ws_size;
  const float* hs = (const float*)d_in[0];   // [B,S,H] fp32
  const float* wp = (const float*)d_in[1];   // [3H,H] fp32
  // d_in[2] = additive causal mask (implemented analytically), d_in[3] = position_ids (= arange)
  float* out = (float*)d_out;

  char* ws = (char*)d_ws;
  __bf16* Xbf = (__bf16*)(ws);                           //  32 MB  [M,K]
  __bf16* Wbf = (__bf16*)(ws + (size_t)33554432);        //  96 MB  [3H,K]
  __bf16* Pbf = (__bf16*)(ws + (size_t)134217728);       //  96 MB  [M,3H]
  // after GEMM, X/W region is dead -> reuse for Q/K/Vt
  __bf16* Qr = (__bf16*)(ws);                            // [B,NH,S,HD]
  __bf16* Kr = (__bf16*)(ws + (size_t)33554432);         // [B,NH,S,HD]
  __bf16* Vt = (__bf16*)(ws + (size_t)67108864);         // [B,NH,HD,S]

  // 1) converts
  cvt_bf16_kernel<<<(MDIM * KDIM / 8 + 255) / 256, 256, 0, stream>>>(hs, Xbf, MDIM * KDIM / 8);
  cvt_bf16_kernel<<<(N3 * KDIM / 8 + 255) / 256, 256, 0, stream>>>(wp, Wbf, N3 * KDIM / 8);

  // 2) QKV GEMM: grid (N/128, M/128)
  dim3 ggrid(N3 / 128, MDIM / 128, 1);
  qkv_gemm_kernel<<<ggrid, 256, 0, stream>>>(Xbf, Wbf, Pbf);

  // 3) RoPE + layout: B*S*NH*64 threads
  rope_kernel<<<(BB * SS * NHD * 64) / 256, 256, 0, stream>>>(Pbf, Qr, Kr, Vt);

  // 4) attention: grid (S/128, NH, B)
  dim3 agrid(SS / 128, NHD, BB);
  attn_kernel<<<agrid, 256, 0, stream>>>(Qr, Kr, Vt, out);
}
